// Qwen3VLVisionModel_51848845197680
// MI455X (gfx1250) — matrix operand order repaired and checked by reading, not compiled, not run on hardware
//
#include <hip/hip_runtime.h>
#include <hip/hip_fp16.h>
#include <hip/hip_bf16.h>

// ---------------- model constants ----------------
#define DEPTH   12
#define DMODEL  1152
#define NH      16
#define HDIM    72
#define HPAD    96          // 72 padded to 3*32 for clean WMMA k-steps
#define ROT     36
#define TOK     4096
#define NSEQ    4
#define SEQL    1024
#define IDIM    4304
#define OUTD    3584
#define D4      4608
#define PIN     1536
#define COLS_OUT (4*OUTD)   // 14336

typedef __attribute__((ext_vector_type(16))) _Float16 v16h;
typedef __attribute__((ext_vector_type(8)))  _Float16 v8h;
typedef __attribute__((ext_vector_type(8)))  float    v8f;
typedef __attribute__((ext_vector_type(4)))  float    v4f;

union AFrag { v16h v; v8h h2[2]; };

__device__ inline v8f v8f_zero() {
  v8f r;
  #pragma unroll
  for (int i = 0; i < 8; ++i) r[i] = 0.0f;
  return r;
}

__device__ inline float gelu_tanh_f(float x) {
  float t = 0.7978845608028654f * (x + 0.044715f * x * x * x);
  return 0.5f * x * (1.0f + tanhf(t));
}
__device__ inline float gelu_erf_f(float x) {
  return 0.5f * x * (1.0f + erff(x * 0.7071067811865475f));
}

// ---------------------------------------------------------------------------
// Generic WMMA GEMM:  C[M,N] = act( A[M,K] * B[N,K]^T + bias[N] (+res) )
// Block tile 128x128, BK=32, 256 threads = 8 waves as 2(M) x 4(N),
// each wave computes 64x32 via 4x2 v_wmma_f32_16x16x32_f16 accumulators.
// A and B are fp32 in global memory, converted to f16 while staging to LDS.
// ---------------------------------------------------------------------------
enum { EPI_NONE = 0, EPI_RES = 1, EPI_GTANH = 2, EPI_GERF = 3, EPI_POS = 4 };

template <int EPI>
__global__ __launch_bounds__(256) void gemm_wmma(
    const float* __restrict__ A, int lda,
    const float* __restrict__ B, int ldb,
    const float* __restrict__ bias,
    float* C, int ldc,
    const float* res,
    const float* __restrict__ pos_table,
    const int*   __restrict__ pos_ids,
    int M, int N, int K)
{
  __shared__ __align__(16) _Float16 sA[128 * 40];
  __shared__ __align__(16) _Float16 sB[128 * 40];

  const int tid   = threadIdx.x;
  const int lane  = tid & 31;
  const int warp  = tid >> 5;
  const int waveM = warp >> 2;      // 0..1
  const int waveN = warp & 3;       // 0..3
  const int m0 = blockIdx.y * 128;
  const int n0 = blockIdx.x * 128;

  v8f acc[4][2];
  #pragma unroll
  for (int i = 0; i < 4; ++i)
    #pragma unroll
    for (int j = 0; j < 2; ++j) acc[i][j] = v8f_zero();

  const int ldrow = tid >> 1;         // 0..127
  const int ldcol = (tid & 1) * 16;   // 0 or 16
  const int am  = lane & 15;
  const int ak  = (lane < 16) ? 0 : 8;   // A chunk select
  const int bk  = (lane < 16) ? 0 : 8;   // B chunk select

  for (int k0 = 0; k0 < K; k0 += 32) {
    __syncthreads();
    // ---- stage A tile (fp32 -> f16) ----
    {
      const int gm = m0 + ldrow;
      union { v8h v[2]; _Float16 e[16]; } ua;
      #pragma unroll
      for (int c = 0; c < 16; c += 4) {
        const int gk = k0 + ldcol + c;
        v4f a = {0.f, 0.f, 0.f, 0.f};
        if (gm < M && gk < K) a = *(const v4f*)(A + (size_t)gm * lda + gk);
        #pragma unroll
        for (int j = 0; j < 4; ++j) ua.e[c + j] = (_Float16)a[j];
      }
      *(v8h*)(sA + ldrow * 40 + ldcol)     = ua.v[0];
      *(v8h*)(sA + ldrow * 40 + ldcol + 8) = ua.v[1];
      if (gm < M && k0 + 64 < K)
        __builtin_prefetch(A + (size_t)gm * lda + k0 + 64, 0, 1);
    }
    // ---- stage B tile (fp32 -> f16) ----
    {
      const int gn = n0 + ldrow;
      union { v8h v[2]; _Float16 e[16]; } ub;
      #pragma unroll
      for (int c = 0; c < 16; c += 4) {
        const int gk = k0 + ldcol + c;
        v4f b = {0.f, 0.f, 0.f, 0.f};
        if (gn < N && gk < K) b = *(const v4f*)(B + (size_t)gn * ldb + gk);
        #pragma unroll
        for (int j = 0; j < 4; ++j) ub.e[c + j] = (_Float16)b[j];
      }
      *(v8h*)(sB + ldrow * 40 + ldcol)     = ub.v[0];
      *(v8h*)(sB + ldrow * 40 + ldcol + 8) = ub.v[1];
      if (gn < N && k0 + 64 < K)
        __builtin_prefetch(B + (size_t)gn * ldb + k0 + 64, 0, 1);
    }
    __syncthreads();

    // ---- fragments + MMA ----
    AFrag af[4];
    #pragma unroll
    for (int mt = 0; mt < 4; ++mt) {
      const _Float16* p = sA + (waveM * 64 + mt * 16 + am) * 40 + ak;
      af[mt].h2[0] = *(const v8h*)p;          // K 0..7  (or 8..15)
      af[mt].h2[1] = *(const v8h*)(p + 16);   // K 16..23 (or 24..31)
    }
    AFrag bf[2];
    #pragma unroll
    for (int nt = 0; nt < 2; ++nt) {
      const _Float16* p = sB + (waveN * 32 + nt * 16 + am) * 40 + bk;
      bf[nt].h2[0] = *(const v8h*)p;          // K 0..7  (or 8..15)
      bf[nt].h2[1] = *(const v8h*)(p + 16);   // K 16..23 (or 24..31)
    }
    #pragma unroll
    for (int mt = 0; mt < 4; ++mt)
      #pragma unroll
      for (int nt = 0; nt < 2; ++nt)
        acc[mt][nt] = __builtin_amdgcn_wmma_f32_16x16x32_f16(
            false, af[mt].v, false, bf[nt].v, (short)0, acc[mt][nt],
            false, false);
  }

  // ---- epilogue ----
  const int hi   = lane >> 4;
  const int colL = lane & 15;
  #pragma unroll
  for (int mt = 0; mt < 4; ++mt)
    #pragma unroll
    for (int nt = 0; nt < 2; ++nt)
      #pragma unroll
      for (int r = 0; r < 8; ++r) {
        const int gm = m0 + waveM * 64 + mt * 16 + r + 8 * hi;
        const int gn = n0 + waveN * 32 + nt * 16 + colL;
        if (gm < M && gn < N) {
          float v = acc[mt][nt][r] + bias[gn];
          if constexpr (EPI == EPI_RES)   v += res[(size_t)gm * ldc + gn];
          if constexpr (EPI == EPI_POS)   v += pos_table[(size_t)pos_ids[gm] * N + gn];
          if constexpr (EPI == EPI_GTANH) v = gelu_tanh_f(v);
          if constexpr (EPI == EPI_GERF)  v = gelu_erf_f(v);
          C[(size_t)gm * ldc + gn] = v;
        }
      }
}

// ---------------------------------------------------------------------------
// LayerNorm over last dim D (one block of 256 threads per row)
// ---------------------------------------------------------------------------
__global__ __launch_bounds__(256) void layernorm_k(
    const float* __restrict__ x, const float* __restrict__ w,
    const float* __restrict__ b, float* __restrict__ y, int D)
{
  const int row = blockIdx.x;
  const float* xr = x + (size_t)row * D;
  __shared__ float red[256];
  float s = 0.f, s2 = 0.f;
  for (int d = threadIdx.x; d < D; d += 256) {
    float v = xr[d]; s += v; s2 += v * v;
  }
  red[threadIdx.x] = s; __syncthreads();
  for (int off = 128; off > 0; off >>= 1) {
    if (threadIdx.x < off) red[threadIdx.x] += red[threadIdx.x + off];
    __syncthreads();
  }
  const float mean = red[0] / (float)D; __syncthreads();
  red[threadIdx.x] = s2; __syncthreads();
  for (int off = 128; off > 0; off >>= 1) {
    if (threadIdx.x < off) red[threadIdx.x] += red[threadIdx.x + off];
    __syncthreads();
  }
  const float var  = red[0] / (float)D - mean * mean;
  const float rstd = rsqrtf(var + 1e-6f);
  float* yr = y + (size_t)row * D;
  for (int d = threadIdx.x; d < D; d += 256)
    yr[d] = (xr[d] - mean) * rstd * w[d] + b[d];
}

// ---------------------------------------------------------------------------
// RoPE + pack q,k,v (fp32 qkv [T,3,H,72]) -> f16 [S,H,L,HPAD] (zero padded)
// ---------------------------------------------------------------------------
__global__ __launch_bounds__(256) void rope_pack_k(
    const float* __restrict__ qkv, const float* __restrict__ cosb,
    const float* __restrict__ sinb,
    _Float16* __restrict__ qh, _Float16* __restrict__ kh,
    _Float16* __restrict__ vh)
{
  const size_t total = (size_t)NSEQ * NH * SEQL * HPAD;
  size_t i = (size_t)blockIdx.x * 256 + threadIdx.x;
  if (i >= total) return;
  const int d   = (int)(i % HPAD);
  size_t r      = i / HPAD;
  const int row = (int)(r % SEQL); r /= SEQL;
  const int h   = (int)(r % NH);
  const int s   = (int)(r / NH);
  const int t   = s * SEQL + row;
  float qv = 0.f, kv = 0.f, vv = 0.f;
  if (d < HDIM) {
    const float* base = qkv + (size_t)t * (3 * DMODEL) + h * HDIM;
    const float q = base[d];
    const float k = base[DMODEL + d];
    vv            = base[2 * DMODEL + d];
    const float cs = cosb[(size_t)t * ROT + (d % ROT)];
    const float sn = sinb[(size_t)t * ROT + (d % ROT)];
    const int   dp = (d < ROT) ? d + ROT : d - ROT;
    const float sg = (d < ROT) ? -1.f : 1.f;
    qv = q * cs + sg * base[dp] * sn;
    kv = k * cs + sg * base[DMODEL + dp] * sn;
  }
  qh[i] = (_Float16)qv; kh[i] = (_Float16)kv; vh[i] = (_Float16)vv;
}

// ---------------------------------------------------------------------------
// Flash attention: one block (4 waves) handles 64 q-rows of one (seq, head).
// Online softmax per wave (16 rows), WMMA for QK^T and P*V.
// ---------------------------------------------------------------------------
__global__ __launch_bounds__(128) void attn_k(
    const _Float16* __restrict__ qh, const _Float16* __restrict__ kh,
    const _Float16* __restrict__ vh, float* __restrict__ o)
{
  __shared__ __align__(16) _Float16 sK[32 * 104];   // [key][feat]
  __shared__ __align__(16) _Float16 sVt[HPAD * 40]; // [hd][key]
  __shared__ __align__(16) _Float16 sP[4 * 16 * 32];

  const int tid  = threadIdx.x;
  const int lane = tid & 31;
  const int wave = tid >> 5;
  int blk = blockIdx.x;
  const int qb = blk % (SEQL / 64); blk /= (SEQL / 64);
  const int h  = blk % NH;
  const int s  = blk / NH;

  const _Float16* qbase = qh + (size_t)(s * NH + h) * SEQL * HPAD;
  const _Float16* kbase = kh + (size_t)(s * NH + h) * SEQL * HPAD;
  const _Float16* vbase = vh + (size_t)(s * NH + h) * SEQL * HPAD;
  const int qrow0 = qb * 64 + wave * 16;

  const int am = lane & 15;
  const int ak = (lane < 16) ? 0 : 8;
  const int bk = (lane < 16) ? 0 : 8;
  const int hi = lane >> 4;
  const int colL = lane & 15;

  AFrag aQ[3];
  #pragma unroll
  for (int ks = 0; ks < 3; ++ks) {
    const _Float16* p = qbase + (size_t)(qrow0 + am) * HPAD + ks * 32 + ak;
    aQ[ks].h2[0] = *(const v8h*)p;
    aQ[ks].h2[1] = *(const v8h*)(p + 16);
  }

  float mrow[8], lrow[8];
  v8f O[6];
  #pragma unroll
  for (int i = 0; i < 8; ++i) { mrow[i] = -1e30f; lrow[i] = 0.f; }
  #pragma unroll
  for (int i = 0; i < 6; ++i) O[i] = v8f_zero();
  const float scale = 0.11785113019775793f; // 1/sqrt(72)

  for (int kt = 0; kt < SEQL / 32; ++kt) {
    __syncthreads();
    // stage K (row-major) and V (transposed) tiles: 32 keys x 96 feats
    #pragma unroll
    for (int j = 0; j < 3; ++j) {
      const int idx = tid + 128 * j;           // 384 8-half chunks
      const int key = idx / 12, ch = idx % 12;
      const size_t g = (size_t)(kt * 32 + key) * HPAD + ch * 8;
      v8h kd = *(const v8h*)(kbase + g);
      *(v8h*)(sK + key * 104 + ch * 8) = kd;
      v8h vd = *(const v8h*)(vbase + g);
      #pragma unroll
      for (int e = 0; e < 8; ++e) sVt[(ch * 8 + e) * 40 + key] = vd[e];
    }
    __syncthreads();

    // S = Q * K^T (16 rows x 32 keys) as two 16x16 frags
    v8f sf[2]; sf[0] = v8f_zero(); sf[1] = v8f_zero();
    #pragma unroll
    for (int ks = 0; ks < 3; ++ks)
      #pragma unroll
      for (int nt = 0; nt < 2; ++nt) {
        AFrag bfK;
        const _Float16* p = sK + (nt * 16 + colL) * 104 + ks * 32 + bk;
        bfK.h2[0] = *(const v8h*)p;
        bfK.h2[1] = *(const v8h*)(p + 16);
        sf[nt] = __builtin_amdgcn_wmma_f32_16x16x32_f16(
            false, aQ[ks].v, false, bfK.v, (short)0, sf[nt], false, false);
      }

    // online softmax update
    #pragma unroll
    for (int r = 0; r < 8; ++r) { sf[0][r] *= scale; sf[1][r] *= scale; }
    float csc[8];
    #pragma unroll
    for (int r = 0; r < 8; ++r) {
      float mx = fmaxf(sf[0][r], sf[1][r]);
      #pragma unroll
      for (int msk = 1; msk < 16; msk <<= 1)
        mx = fmaxf(mx, __shfl_xor(mx, msk, 32));
      const float mnew = fmaxf(mrow[r], mx);
      csc[r] = expf(mrow[r] - mnew);
      mrow[r] = mnew;
      const float p0 = expf(sf[0][r] - mnew);
      const float p1 = expf(sf[1][r] - mnew);
      float ps = p0 + p1;
      #pragma unroll
      for (int msk = 1; msk < 16; msk <<= 1) ps += __shfl_xor(ps, msk, 32);
      lrow[r] = lrow[r] * csc[r] + ps;
      _Float16* pw = sP + wave * 512 + (r + 8 * hi) * 32;
      pw[colL]      = (_Float16)p0;
      pw[16 + colL] = (_Float16)p1;
    }
    #pragma unroll
    for (int i = 0; i < 6; ++i)
      #pragma unroll
      for (int r = 0; r < 8; ++r) O[i][r] *= csc[r];

    asm volatile("s_wait_dscnt 0" ::: "memory"); // P stores visible to wave

    AFrag aP;
    {
      const _Float16* pp = sP + wave * 512 + am * 32 + ak;
      aP.h2[0] = *(const v8h*)pp;
      aP.h2[1] = *(const v8h*)(pp + 16);
    }
    #pragma unroll
    for (int nt = 0; nt < 6; ++nt) {
      AFrag bfV;
      const _Float16* pv = sVt + (nt * 16 + colL) * 40 + bk;
      bfV.h2[0] = *(const v8h*)pv;
      bfV.h2[1] = *(const v8h*)(pv + 16);
      O[nt] = __builtin_amdgcn_wmma_f32_16x16x32_f16(
          false, aP.v, false, bfV.v, (short)0, O[nt], false, false);
    }
  }

  // write O / l  ->  o[token, h*72+hd]
  #pragma unroll
  for (int nt = 0; nt < 6; ++nt) {
    const int hd = nt * 16 + colL;
    #pragma unroll
    for (int r = 0; r < 8; ++r) {
      if (hd < HDIM) {
        const int token = s * SEQL + qrow0 + r + 8 * hi;
        o[(size_t)token * DMODEL + h * HDIM + hd] = O[nt][r] / lrow[r];
      }
    }
  }
}

// ---------------------------------------------------------------------------
// Host-side launch
// ---------------------------------------------------------------------------
static inline void launch_gemm(int epi, const float* A, int lda,
                               const float* B, int ldb, const float* bias,
                               float* C, int ldc, const float* res,
                               const float* pt, const int* pi,
                               int M, int N, int K, hipStream_t stream)
{
  dim3 g((N + 127) / 128, (M + 127) / 128), blk(256);
  switch (epi) {
    case EPI_NONE:  gemm_wmma<EPI_NONE ><<<g, blk, 0, stream>>>(A, lda, B, ldb, bias, C, ldc, res, pt, pi, M, N, K); break;
    case EPI_RES:   gemm_wmma<EPI_RES  ><<<g, blk, 0, stream>>>(A, lda, B, ldb, bias, C, ldc, res, pt, pi, M, N, K); break;
    case EPI_GTANH: gemm_wmma<EPI_GTANH><<<g, blk, 0, stream>>>(A, lda, B, ldb, bias, C, ldc, res, pt, pi, M, N, K); break;
    case EPI_GERF:  gemm_wmma<EPI_GERF ><<<g, blk, 0, stream>>>(A, lda, B, ldb, bias, C, ldc, res, pt, pi, M, N, K); break;
    case EPI_POS:   gemm_wmma<EPI_POS  ><<<g, blk, 0, stream>>>(A, lda, B, ldb, bias, C, ldc, res, pt, pi, M, N, K); break;
  }
}

extern "C" void kernel_launch(void* const* d_in, const int* in_sizes, int n_in,
                              void* d_out, int out_size, void* d_ws, size_t ws_size,
                              hipStream_t stream)
{
  (void)in_sizes; (void)n_in; (void)out_size; (void)ws_size;
  const float* patches   = (const float*)d_in[0];
  const int*   pos_ids   = (const int*)  d_in[1];
  const float* cosb      = (const float*)d_in[3];
  const float* sinb      = (const float*)d_in[4];
  const float* patch_w   = (const float*)d_in[5];
  const float* patch_b   = (const float*)d_in[6];
  const float* pos_table = (const float*)d_in[7];
  const float* ln1_w     = (const float*)d_in[8];
  const float* ln1_b     = (const float*)d_in[9];
  const float* qkv_w     = (const float*)d_in[10];
  const float* qkv_b     = (const float*)d_in[11];
  const float* outp_w    = (const float*)d_in[12];
  const float* outp_b    = (const float*)d_in[13];
  const float* ln2_w     = (const float*)d_in[14];
  const float* ln2_b     = (const float*)d_in[15];
  const float* fc1_w     = (const float*)d_in[16];
  const float* fc1_b     = (const float*)d_in[17];
  const float* fc2_w     = (const float*)d_in[18];
  const float* fc2_b     = (const float*)d_in[19];
  const float* m_ln_w    = (const float*)d_in[20];
  const float* m_ln_b    = (const float*)d_in[21];
  const float* m_fc1_w   = (const float*)d_in[22];
  const float* m_fc1_b   = (const float*)d_in[23];
  const float* m_fc2_w   = (const float*)d_in[24];
  const float* m_fc2_b   = (const float*)d_in[25];
  const float* ds_ln_w   = (const float*)d_in[26];
  const float* ds_ln_b   = (const float*)d_in[27];
  const float* ds_fc1_w  = (const float*)d_in[28];
  const float* ds_fc1_b  = (const float*)d_in[29];
  const float* ds_fc2_w  = (const float*)d_in[30];
  const float* ds_fc2_b  = (const float*)d_in[31];
  float* out = (float*)d_out;

  // workspace carve-up
  char* ws = (char*)d_ws;
  size_t off = 0;
  auto alloc = [&](size_t bytes) -> void* {
    void* p = ws + off; off += (bytes + 255) & ~(size_t)255; return p;
  };
  float*     x    = (float*)alloc((size_t)TOK * DMODEL * 4);
  float*     hbuf = (float*)alloc((size_t)TOK * DMODEL * 4);
  float*     qkv  = (float*)alloc((size_t)TOK * 3 * DMODEL * 4);
  _Float16*  qh   = (_Float16*)alloc((size_t)NSEQ * NH * SEQL * HPAD * 2);
  _Float16*  kh   = (_Float16*)alloc((size_t)NSEQ * NH * SEQL * HPAD * 2);
  _Float16*  vh   = (_Float16*)alloc((size_t)NSEQ * NH * SEQL * HPAD * 2);
  float*     obuf = (float*)alloc((size_t)TOK * DMODEL * 4);
  float*     mlp  = (float*)alloc((size_t)TOK * IDIM * 4);
  float*     dsb0 = (float*)alloc((size_t)TOK * DMODEL * 4);
  float*     dsb1 = (float*)alloc((size_t)TOK * DMODEL * 4);
  float*     dsb2 = (float*)alloc((size_t)TOK * DMODEL * 4);
  float*     dsb[3] = { dsb0, dsb1, dsb2 };

  // ---- patch embed: x = patches @ patch_w^T + patch_b + pos_table[pos_ids] ----
  launch_gemm(EPI_POS, patches, PIN, patch_w, PIN, patch_b, x, DMODEL,
              nullptr, pos_table, pos_ids, TOK, DMODEL, PIN, stream);

  const size_t ropeTotal = (size_t)NSEQ * NH * SEQL * HPAD;
  int dsn = 0;
  for (int i = 0; i < DEPTH; ++i) {
    // attn block
    layernorm_k<<<TOK, 256, 0, stream>>>(x, ln1_w + (size_t)i * DMODEL,
                                         ln1_b + (size_t)i * DMODEL, hbuf, DMODEL);
    launch_gemm(EPI_NONE, hbuf, DMODEL, qkv_w + (size_t)i * 3 * DMODEL * DMODEL,
                DMODEL, qkv_b + (size_t)i * 3 * DMODEL, qkv, 3 * DMODEL,
                nullptr, nullptr, nullptr, TOK, 3 * DMODEL, DMODEL, stream);
    rope_pack_k<<<(unsigned)((ropeTotal + 255) / 256), 256, 0, stream>>>(
        qkv, cosb, sinb, qh, kh, vh);
    attn_k<<<NSEQ * NH * (SEQL / 64), 128, 0, stream>>>(qh, kh, vh, obuf);
    launch_gemm(EPI_RES, obuf, DMODEL, outp_w + (size_t)i * DMODEL * DMODEL,
                DMODEL, outp_b + (size_t)i * DMODEL, x, DMODEL,
                x, nullptr, nullptr, TOK, DMODEL, DMODEL, stream);
    // mlp block
    layernorm_k<<<TOK, 256, 0, stream>>>(x, ln2_w + (size_t)i * DMODEL,
                                         ln2_b + (size_t)i * DMODEL, hbuf, DMODEL);
    launch_gemm(EPI_GTANH, hbuf, DMODEL, fc1_w + (size_t)i * IDIM * DMODEL,
                DMODEL, fc1_b + (size_t)i * IDIM, mlp, IDIM,
                nullptr, nullptr, nullptr, TOK, IDIM, DMODEL, stream);
    launch_gemm(EPI_RES, mlp, IDIM, fc2_w + (size_t)i * DMODEL * IDIM,
                IDIM, fc2_b + (size_t)i * DMODEL, x, DMODEL,
                x, nullptr, nullptr, TOK, DMODEL, IDIM, stream);
    if (i == 3 || i == 6 || i == 9) {
      (void)hipMemcpyAsync(dsb[dsn], x, (size_t)TOK * DMODEL * 4,
                           hipMemcpyDeviceToDevice, stream);
      ++dsn;
    }
  }

  // ---- merger main branch ----
  const int MM = TOK * DMODEL / D4;   // 1024 rows after reshape
  layernorm_k<<<TOK, 256, 0, stream>>>(x, m_ln_w, m_ln_b, hbuf, DMODEL);
  launch_gemm(EPI_GERF, hbuf, D4, m_fc1_w, D4, m_fc1_b, obuf, D4,
              nullptr, nullptr, nullptr, MM, D4, D4, stream);
  launch_gemm(EPI_NONE, obuf, D4, m_fc2_w, D4, m_fc2_b, out, COLS_OUT,
              nullptr, nullptr, nullptr, MM, OUTD, D4, stream);

  // ---- deepstack branches (LN over 4608) ----
  for (int j = 0; j < 3; ++j) {
    layernorm_k<<<MM, 256, 0, stream>>>(dsb[j], ds_ln_w + (size_t)j * D4,
                                        ds_ln_b + (size_t)j * D4, hbuf, D4);
    launch_gemm(EPI_GERF, hbuf, D4, ds_fc1_w + (size_t)j * D4 * D4, D4,
                ds_fc1_b + (size_t)j * D4, obuf, D4,
                nullptr, nullptr, nullptr, MM, D4, D4, stream);
    launch_gemm(EPI_NONE, obuf, D4, ds_fc2_w + (size_t)j * OUTD * D4, D4,
                ds_fc2_b + (size_t)j * OUTD, out + (size_t)(j + 1) * OUTD,
                COLS_OUT, nullptr, nullptr, nullptr, MM, OUTD, D4, stream);
  }
}
